// MoELayer_49641232007623
// MI455X (gfx1250) — matrix operand order repaired and checked by reading, not compiled, not run on hardware
//
#include <hip/hip_runtime.h>

#define D_MODEL   1024
#define D_INNER   2048
#define N_EXPERTS 8
#define NTOK      8192                 // 4 * 2048 tokens
#define OUT_ELEMS (NTOK * D_MODEL)     // 8388608; d_out also holds aux at [OUT_ELEMS]

typedef __attribute__((ext_vector_type(16))) __bf16 v16bf;
typedef __attribute__((ext_vector_type(8)))  __bf16 v8bf;
typedef __attribute__((ext_vector_type(4)))  __bf16 v4bf;
typedef __attribute__((ext_vector_type(8)))  float  v8f;

// ---------------------------------------------------------------- helpers ----

__device__ __forceinline__ float silu_f(float v) {
    return v * __builtin_amdgcn_rcpf(1.0f + __expf(-v));
}

__device__ __forceinline__ unsigned lds_off(const void* p) {
    return (unsigned)(unsigned long long)p;
}

// CDNA5 async copy: per-lane 16B global -> LDS, tracked by ASYNCcnt.
__device__ __forceinline__ void async_g2l_b128(unsigned ldsoff, const void* g) {
    asm volatile("global_load_async_to_lds_b128 %0, %1, off"
                 :: "v"(ldsoff), "v"((unsigned long long)g)
                 : "memory");
}
__device__ __forceinline__ void wait_async0() {
    asm volatile("s_wait_asynccnt 0" ::: "memory");
}

__device__ __forceinline__ v16bf frag16(v8bf lo, v8bf hi) {
    v16bf o;
#pragma unroll
    for (int i = 0; i < 8; ++i) { o[i] = lo[i]; o[i + 8] = hi[i]; }
    return o;
}

// A fragment 16x32 bf16 from LDS tile [rows][32], row m.
// ISA layout: lanes 0-15 K={0..7,16..23}, lanes 16-31 K={8..15,24..31}.
__device__ __forceinline__ v16bf lds_frag_a(const __bf16* sA, int m, int half) {
    const __bf16* r = sA + m * 32 + 8 * half;
    return frag16(*(const v8bf*)(r + 0), *(const v8bf*)(r + 16));
}

// B fragment 32x16 bf16 from LDS tile [rows][32], row n; lanes 0-15 K={0..7,16..23}, lanes 16-31 K={8..15,24..31}.
__device__ __forceinline__ v16bf lds_frag_b(const __bf16* sB, int n, int half) {
    const __bf16* r = sB + n * 32 + 8 * half;
    return frag16(*(const v8bf*)(r + 0), *(const v8bf*)(r + 16));
}

#define WMMA_BF16(A, B, C) \
    __builtin_amdgcn_wmma_f32_16x16x32_bf16(false, (A), false, (B), (short)0, (C), false, false)

// ---------------------------------------------------------------- kernels ----

__global__ void __launch_bounds__(256) moe_init(float* __restrict__ out,
                                                unsigned* __restrict__ counts) {
    long i = (long)blockIdx.x * blockDim.x + threadIdx.x;
    if (i <= OUT_ELEMS) out[i] = 0.0f;      // includes aux slot
    if (i < N_EXPERTS)  counts[i] = 0u;
}

__global__ void __launch_bounds__(256) cvt_bf16(const float* __restrict__ s,
                                                __bf16* __restrict__ d, long n) {
    long i = ((long)blockIdx.x * 256 + threadIdx.x) * 4;
    if (i >= n) return;
    float4 v = *(const float4*)(s + i);
    v4bf o;
    o[0] = (__bf16)v.x; o[1] = (__bf16)v.y; o[2] = (__bf16)v.z; o[3] = (__bf16)v.w;
    *(v4bf*)(d + i) = o;
}

// Router: softmax -> top-2 -> gate weights cw[t][e]; builds per-expert compacted
// token index lists (atomic append; list order does not change the final math).
__global__ void __launch_bounds__(256) moe_router(const float* __restrict__ x,
                                                  const float* __restrict__ Wr,
                                                  const float* __restrict__ br,
                                                  float* __restrict__ cw,
                                                  unsigned* __restrict__ counts,
                                                  int* __restrict__ idx) {
    int t = blockIdx.x * 256 + threadIdx.x;
    if (t >= NTOK) return;
    float acc[N_EXPERTS];
#pragma unroll
    for (int e = 0; e < N_EXPERTS; ++e) acc[e] = br[e];
    const float* xr = x + (long)t * D_MODEL;
    for (int c = 0; c < D_MODEL; c += 4) {
        float4 xv = *(const float4*)(xr + c);
#pragma unroll
        for (int e = 0; e < N_EXPERTS; ++e) {
            const float* w = Wr + (long)e * D_MODEL + c;
            acc[e] += xv.x * w[0] + xv.y * w[1] + xv.z * w[2] + xv.w * w[3];
        }
    }
    float m = acc[0];
#pragma unroll
    for (int e = 1; e < N_EXPERTS; ++e) m = fmaxf(m, acc[e]);
    float p[N_EXPERTS], s = 0.0f;
#pragma unroll
    for (int e = 0; e < N_EXPERTS; ++e) { p[e] = __expf(acc[e] - m); s += p[e]; }
    float inv = __builtin_amdgcn_rcpf(s);
#pragma unroll
    for (int e = 0; e < N_EXPERTS; ++e) p[e] *= inv;
    int bi = 0; float bv = p[0];
#pragma unroll
    for (int e = 1; e < N_EXPERTS; ++e) if (p[e] > bv) { bv = p[e]; bi = e; }
    int si = -1; float sv = -1.0f;
#pragma unroll
    for (int e = 0; e < N_EXPERTS; ++e)
        if (e != bi && p[e] > sv) { sv = p[e]; si = e; }
    float rw = __builtin_amdgcn_rcpf(bv + sv);
#pragma unroll
    for (int e = 0; e < N_EXPERTS; ++e) cw[(long)t * N_EXPERTS + e] = 0.0f;
    cw[(long)t * N_EXPERTS + bi] = bv * rw;
    cw[(long)t * N_EXPERTS + si] = sv * rw;
    unsigned p0 = atomicAdd(&counts[bi], 1u);
    idx[bi * NTOK + p0] = t;
    unsigned p1 = atomicAdd(&counts[si], 1u);
    idx[si * NTOK + p1] = t;
}

__global__ void moe_aux(const unsigned* __restrict__ counts, float* __restrict__ out) {
    float aux = 0.0f;
#pragma unroll
    for (int e = 0; e < N_EXPERTS; ++e) {
        float l = (float)counts[e] / (float)NTOK;
        aux += l * l;
    }
    out[OUT_ELEMS] = aux;
}

// Stage A (per expert): compacted tokens, both halves of x @ Win[j].T with fused
// SiLU-gate activation -> compacted bf16 rows yc.
// Block: 8 waves, tile 128(rows) x 32(channels).
// Double buffered, K=64 per segment (2 slabs of K=32) -> 8 WMMA per sync.
__global__ void __launch_bounds__(256) moe_up_act(const __bf16* __restrict__ xb,
                                                  const __bf16* __restrict__ winb,
                                                  const float* __restrict__ convw_j,
                                                  const float* __restrict__ convb_j,
                                                  const float* __restrict__ Dp_j,
                                                  const unsigned* __restrict__ counts,
                                                  const int* __restrict__ idx_j,
                                                  __bf16* __restrict__ yc, int j) {
    const int cnt = (int)counts[j];
    const int t0  = blockIdx.x * 128;
    if (t0 >= cnt) return;
    const int d0 = blockIdx.y * 32;

    // 4 slabs = 2 buffers x 2 K-substeps, each slab [rows][32]
    __shared__ __attribute__((aligned(16))) __bf16 sA [4][128 * 32];
    __shared__ __attribute__((aligned(16))) __bf16 sBi[4][32 * 32];
    __shared__ __attribute__((aligned(16))) __bf16 sBz[4][32 * 32];

    const unsigned tid = threadIdx.x;
    // A staging: 512 x 16B chunks per slab, 2 per thread
    int r0 = tid >> 2,          col0 = (tid & 3) * 8;
    int r1 = (tid + 256) >> 2,  col1 = (tid & 3) * 8;
    int q0 = t0 + r0; if (q0 > cnt - 1) q0 = cnt - 1;   // clamp: duplicate last row
    int q1 = t0 + r1; if (q1 > cnt - 1) q1 = cnt - 1;
    const __bf16* aG0 = xb + (long)idx_j[q0] * D_MODEL + col0;
    const __bf16* aG1 = xb + (long)idx_j[q1] * D_MODEL + col1;
    unsigned aL0 = lds_off(&sA[0][0]) + (unsigned)(r0 * 32 + col0) * 2u;
    unsigned aL1 = lds_off(&sA[0][0]) + (unsigned)(r1 * 32 + col1) * 2u;
    // B staging: waves 0-3 -> Bi, waves 4-7 -> Bz (128 chunks each per slab)
    int cb = tid & 127;
    int br_ = cb >> 2, bc_ = (cb & 3) * 8;
    bool isZ = tid >= 128;
    const __bf16* bG = winb + (long)(d0 + br_ + (isZ ? D_INNER : 0)) * D_MODEL + bc_;
    unsigned bL = lds_off(isZ ? &sBz[0][0] : &sBi[0][0]) + (unsigned)(br_ * 32 + bc_) * 2u;

    const int lane = threadIdx.x & 31;
    const int half = lane >> 4;
    const int nl   = lane & 15;
    const int wid  = threadIdx.x >> 5;
    const int m_off = (wid & 3) * 32;       // wave: 32 rows (2 A-frags) x 16 cols
    const int n_off = (wid >> 2) * 16;

    v8f ai0 = {}, ai1 = {}, az0 = {}, az1 = {};

    auto stage = [&](int buf, int k) {      // stage K=64 (2 slabs) into buffer
#pragma unroll
        for (int s = 0; s < 2; ++s) {
            unsigned sl = (unsigned)(2 * buf + s);
            async_g2l_b128(aL0 + sl * 8192u, aG0 + k + s * 32);
            async_g2l_b128(aL1 + sl * 8192u, aG1 + k + s * 32);
            async_g2l_b128(bL + sl * 2048u, bG + k + s * 32);
        }
    };
    auto compute = [&](int buf) {           // 2 slabs -> 8 WMMA
#pragma unroll
        for (int s = 0; s < 2; ++s) {
            int sl = 2 * buf + s;
            v16bf a0 = lds_frag_a(sA[sl], m_off + nl, half);
            v16bf a1 = lds_frag_a(sA[sl], m_off + 16 + nl, half);
            v16bf fi = lds_frag_b(sBi[sl], n_off + nl, half);
            v16bf fz = lds_frag_b(sBz[sl], n_off + nl, half);
            ai0 = WMMA_BF16(a0, fi, ai0);
            ai1 = WMMA_BF16(a1, fi, ai1);
            az0 = WMMA_BF16(a0, fz, az0);
            az1 = WMMA_BF16(a1, fz, az1);
        }
    };

    stage(0, 0);
    wait_async0();
    __syncthreads();
    const int nseg = D_MODEL / 64;          // 16 (even)
    int ks = 0;
    for (; ks + 2 < nseg; ks += 2) {        // branch-free pipelined main loop
        stage(1, (ks + 1) * 64);
        compute(0);
        wait_async0(); __syncthreads();
        stage(0, (ks + 2) * 64);
        compute(1);
        wait_async0(); __syncthreads();
    }
    stage(1, (ks + 1) * 64);                // tail: last two segments
    compute(0);
    wait_async0(); __syncthreads();
    compute(1);

    int d = d0 + n_off + nl;
    float w3 = convw_j[d * 4 + 3];          // only the last conv tap matters
    float bb = convb_j[d];
    float dp = Dp_j[d];
#pragma unroll
    for (int f = 0; f < 2; ++f) {
        v8f xi = f ? ai1 : ai0;
        v8f zz = f ? az1 : az0;
#pragma unroll
        for (int v = 0; v < 8; ++v) {
            int p = t0 + m_off + f * 16 + v + 8 * half;   // compacted row (< 8192)
            float yv = silu_f(xi[v] * w3 + bb) * dp * silu_f(zz[v]);
            yc[(long)p * D_INNER + d] = (__bf16)yv;
        }
    }
}

// Stage B (per expert): out[idx[p]] += cw * (yc @ Wout[j].T) over compacted rows.
// Block: 8 waves, tile 128(rows) x 64(channels); wave = 32x32.
// Double buffered, K=64 per segment -> 8 WMMA per sync.
__global__ void __launch_bounds__(256) moe_down(const __bf16* __restrict__ yc,
                                                const __bf16* __restrict__ woutb,
                                                const float* __restrict__ cw,
                                                const unsigned* __restrict__ counts,
                                                const int* __restrict__ idx_j,
                                                float* __restrict__ out, int j) {
    const int cnt = (int)counts[j];
    const int t0  = blockIdx.x * 128;
    if (t0 >= cnt) return;
    const int c0 = blockIdx.y * 64;

    __shared__ __attribute__((aligned(16))) __bf16 sA[4][128 * 32];
    __shared__ __attribute__((aligned(16))) __bf16 sB[4][64 * 32];

    const unsigned tid = threadIdx.x;
    int r0 = tid >> 2,          col0 = (tid & 3) * 8;
    int r1 = (tid + 256) >> 2,  col1 = (tid & 3) * 8;
    const __bf16* aG0 = yc + (long)(t0 + r0) * D_INNER + col0;   // compacted rows
    const __bf16* aG1 = yc + (long)(t0 + r1) * D_INNER + col1;
    unsigned aL0 = lds_off(&sA[0][0]) + (unsigned)(r0 * 32 + col0) * 2u;
    unsigned aL1 = lds_off(&sA[0][0]) + (unsigned)(r1 * 32 + col1) * 2u;
    int br_ = tid >> 2, bc_ = (tid & 3) * 8;                     // 256 B-chunks/slab
    const __bf16* bG = woutb + (long)(c0 + br_) * D_INNER + bc_;
    unsigned bL = lds_off(&sB[0][0]) + (unsigned)(br_ * 32 + bc_) * 2u;

    const int lane = threadIdx.x & 31;
    const int half = lane >> 4;
    const int nl   = lane & 15;
    const int wid  = threadIdx.x >> 5;
    const int m_off = (wid & 3) * 32;       // wave: 32 rows x 32 cols
    const int n_off = (wid >> 2) * 32;

    v8f d00 = {}, d10 = {}, d01 = {}, d11 = {};

    auto stage = [&](int buf, int k) {
#pragma unroll
        for (int s = 0; s < 2; ++s) {
            unsigned sl = (unsigned)(2 * buf + s);
            async_g2l_b128(aL0 + sl * 8192u, aG0 + k + s * 32);
            async_g2l_b128(aL1 + sl * 8192u, aG1 + k + s * 32);
            async_g2l_b128(bL + sl * 4096u, bG + k + s * 32);
        }
    };
    auto compute = [&](int buf) {
#pragma unroll
        for (int s = 0; s < 2; ++s) {
            int sl = 2 * buf + s;
            v16bf a0 = lds_frag_a(sA[sl], m_off + nl, half);
            v16bf a1 = lds_frag_a(sA[sl], m_off + 16 + nl, half);
            v16bf b0 = lds_frag_b(sB[sl], n_off + nl, half);
            v16bf b1 = lds_frag_b(sB[sl], n_off + 16 + nl, half);
            d00 = WMMA_BF16(a0, b0, d00);
            d10 = WMMA_BF16(a1, b0, d10);
            d01 = WMMA_BF16(a0, b1, d01);
            d11 = WMMA_BF16(a1, b1, d11);
        }
    };

    stage(0, 0);
    wait_async0();
    __syncthreads();
    const int nseg = D_INNER / 64;          // 32 (even)
    int ks = 0;
    for (; ks + 2 < nseg; ks += 2) {
        stage(1, (ks + 1) * 64);
        compute(0);
        wait_async0(); __syncthreads();
        stage(0, (ks + 2) * 64);
        compute(1);
        wait_async0(); __syncthreads();
    }
    stage(1, (ks + 1) * 64);
    compute(0);
    wait_async0(); __syncthreads();
    compute(1);

#pragma unroll
    for (int g = 0; g < 2; ++g) {
        int c = c0 + n_off + g * 16 + nl;
#pragma unroll
        for (int f = 0; f < 2; ++f) {
            v8f acc = g ? (f ? d11 : d01) : (f ? d10 : d00);
#pragma unroll
            for (int v = 0; v < 8; ++v) {
                int p = t0 + m_off + f * 16 + v + 8 * half;
                if (p < cnt) {              // tail rows of last tile are padding
                    int t = idx_j[p];
                    out[(long)t * D_MODEL + c] += cw[(long)t * N_EXPERTS + j] * acc[v];
                }
            }
        }
    }
}

// ----------------------------------------------------------------- launch ----

extern "C" void kernel_launch(void* const* d_in, const int* in_sizes, int n_in,
                              void* d_out, int out_size, void* d_ws, size_t ws_size,
                              hipStream_t stream) {
    const float* x      = (const float*)d_in[0];   // (4,2048,1024)
    const float* Wr     = (const float*)d_in[1];   // (8,1024)
    const float* br     = (const float*)d_in[2];   // (8,)
    const float* Win    = (const float*)d_in[3];   // (8,4096,1024)
    const float* conv_w = (const float*)d_in[4];   // (8,2048,4)
    const float* conv_b = (const float*)d_in[5];   // (8,2048)
    const float* Dp     = (const float*)d_in[6];   // (8,2048)
    const float* Wout   = (const float*)d_in[7];   // (8,1024,2048)
    float* out = (float*)d_out;

    // workspace layout (~60.5 MB)
    char* ws = (char*)d_ws;
    float*    cw     = (float*)ws;                       // 8192*8 f32      (262144 B)
    unsigned* counts = (unsigned*)(ws + 262144);         // 8 u32
    int*      idx    = (int*)(ws + 262208);              // 8*8192 i32     (262144 B)
    __bf16*   xb     = (__bf16*)(ws + 524352);           // 8192x1024 bf16
    __bf16*   yc     = (__bf16*)(ws + 17301568);         // 8192x2048 bf16 (compacted)
    __bf16*   winb   = (__bf16*)(ws + 50856000);         // 4096x1024 bf16 (per-expert)
    __bf16*   woutb  = (__bf16*)(ws + 59244608);         // 1024x2048 bf16 (per-expert)

    moe_init<<<(OUT_ELEMS + 1 + 255) / 256, 256, 0, stream>>>(out, counts);
    moe_router<<<NTOK / 256, 256, 0, stream>>>(x, Wr, br, cw, counts, idx);
    moe_aux<<<1, 1, 0, stream>>>(counts, out);
    cvt_bf16<<<(NTOK * D_MODEL) / 1024, 256, 0, stream>>>(x, xb, (long)NTOK * D_MODEL);

    for (int j = 0; j < N_EXPERTS; ++j) {
        const long win_n  = (long)2 * D_INNER * D_MODEL;   // 4M elements
        const long wout_n = (long)D_MODEL * D_INNER;       // 2M elements
        cvt_bf16<<<win_n / 1024, 256, 0, stream>>>(Win + (size_t)j * win_n, winb, win_n);
        cvt_bf16<<<wout_n / 1024, 256, 0, stream>>>(Wout + (size_t)j * wout_n, woutb, wout_n);
        moe_up_act<<<dim3(NTOK / 128, D_INNER / 32), 256, 0, stream>>>(
            xb, winb,
            conv_w + (size_t)j * D_INNER * 4,
            conv_b + (size_t)j * D_INNER,
            Dp     + (size_t)j * D_INNER,
            counts, idx + (size_t)j * NTOK, yc, j);
        moe_down<<<dim3(NTOK / 128, D_MODEL / 64), 256, 0, stream>>>(
            yc, woutb, cw, counts, idx + (size_t)j * NTOK, out, j);
    }
}
